// GraphAttnConv_80865644249291
// MI455X (gfx1250) — matrix operand order repaired and checked by reading, not compiled, not run on hardware
//
#include <hip/hip_runtime.h>
#include <hip/hip_bf16.h>

typedef __bf16 bf16;
typedef __attribute__((ext_vector_type(16))) __bf16 v16bf;
typedef __attribute__((ext_vector_type(8)))  __bf16 v8bf;
typedef __attribute__((ext_vector_type(8)))  float  v8f;

#define O_N   50000
#define T_N   100000
#define DIN_N 128
#define H_N   512
#define DOUT_N 128
#define K1_N  (3*DIN_N)        // 384
#define N2_N  (2*H_N+DOUT_N)   // 1152

// ---------------------------------------------------------------------------
// CDNA5 primitives
// ---------------------------------------------------------------------------
__device__ __forceinline__ v8f wmma_bf16(v16bf a, v16bf b, v8f c) {
  return __builtin_amdgcn_wmma_f32_16x16x32_bf16(false, a, false, b,
                                                 (short)0, c, false, false);
}

// Async copy 16B/lane global -> LDS, tracked with ASYNCcnt (gfx1250).
// LDS byte offset = low 32 bits of the flat shared-pointer address.
__device__ __forceinline__ void async_ld_b128(bf16* lds_dst, const bf16* gsrc) {
  unsigned l = (unsigned)(size_t)lds_dst;
  asm volatile("global_load_async_to_lds_b128 %0, %1, off"
               :: "v"(l), "v"(gsrc) : "memory");
}

template<int N>
__device__ __forceinline__ void wait_async() {
  asm volatile("s_wait_asynccnt %0" :: "n"(N) : "memory");
}

struct GemmParams {
  const bf16*  A;      // AMODE 0,2 : bf16 activations (row-major, stride lda)
  const float* Af;     // AMODE 3   : f32 activations
  const float* obj;    // AMODE 1   : obj_vecs (f32, O x 128)
  const float* pred;   // AMODE 1   : pred_vecs (f32, T x 128)
  const int*   edges;  // AMODE 1   : (T,2) int32
  const bf16*  Bt;     // weights transposed: (N,K) bf16 row-major
  const float* bias;   // f32[N] or null
  int M, N, K, lda;
  int relu;
  float* outf;         // optional f32 out (M,N)
  bf16*  outb;         // optional bf16 out (M,N)
  float* outw;         // optional f32 column-window out
  int wlo, whi, wld;
  // DOT epilogue (scores):
  const float* prevp;  // (n_obj, N) f32
  const int*   cidx;   // per-row segment index
  float*       scores; // atomicAdd target, len M
  int tsplit, candoff; // AMODE 2 row split / col offset
};

// Tile: 128(M) x 128(N), K-step 32, double-buffered LDS with async staging.
// 256 threads = 8 waves (wave32): 4(M) x 2(N); each wave does 32x64 via
// 2x4 fragments of v_wmma_f32_16x16x32_bf16.
template<int AMODE, bool DOT>
__global__ __launch_bounds__(256)
void wmma_gemm(GemmParams p) {
  __shared__ bf16 As[2][128 * 40];  // stride 40 halves -> conflict-free frag reads
  __shared__ bf16 Bs[2][128 * 40];

  const int m0 = blockIdx.y * 128;
  const int n0 = blockIdx.x * 128;
  const int tid  = threadIdx.x;
  const int wid  = tid >> 5;
  const int lane = tid & 31;
  const int wm = (wid & 3) * 32;
  const int wn = (wid >> 2) * 64;
  const int lrow  = lane & 15;
  const int lhalf = lane >> 4;

  const int lr = tid >> 1;        // tile row (A) / Bt row (B): 0..127
  const int kh = (tid & 1) * 16;  // 0 or 16 within the 32-wide K slice

  // async instructions issued per wave per stage
  constexpr int OPS = (AMODE == 0 || AMODE == 2) ? 4 : 2;

  v8f acc[2][4] = {};

  auto stage = [&](int t, int buf) {
    const int k0 = t * 32;
    const int gr = m0 + lr;
    const int cr = (gr < p.M) ? gr : (p.M - 1);  // clamp; OOR rows discarded in epilogue
    bf16* dstA = &As[buf][lr * 40 + kh];
    if (AMODE == 0) {
      const bf16* s = p.A + (size_t)cr * p.lda + k0 + kh;
      async_ld_b128(dstA, s);
      async_ld_b128(dstA + 8, s + 8);
    } else if (AMODE == 2) {
      const bf16* base = (cr < p.tsplit)
          ? (p.A + (size_t)cr * p.lda)
          : (p.A + (size_t)(cr - p.tsplit) * p.lda + p.candoff);
      const bf16* s = base + k0 + kh;
      async_ld_b128(dstA, s);
      async_ld_b128(dstA + 8, s + 8);
    } else if (AMODE == 3) {
      const float4* s = (const float4*)(p.Af + (size_t)cr * p.lda + k0 + kh);
      float tf[16];
      ((float4*)tf)[0] = s[0]; ((float4*)tf)[1] = s[1];
      ((float4*)tf)[2] = s[2]; ((float4*)tf)[3] = s[3];
      #pragma unroll
      for (int e = 0; e < 16; ++e) dstA[e] = (bf16)tf[e];
    } else { // AMODE 1: fused gather-concat [obj[s], pred, obj[o]]
      const int kk = k0 + kh;      // 16-chunk never crosses a 128 boundary
      const float* s;
      if (kk < DIN_N)        s = p.obj  + (size_t)p.edges[2*cr]   * DIN_N + kk;
      else if (kk < 2*DIN_N) s = p.pred + (size_t)cr              * DIN_N + (kk - DIN_N);
      else                   s = p.obj  + (size_t)p.edges[2*cr+1] * DIN_N + (kk - 2*DIN_N);
      const float4* s4 = (const float4*)s;
      float tf[16];
      ((float4*)tf)[0] = s4[0]; ((float4*)tf)[1] = s4[1];
      ((float4*)tf)[2] = s4[2]; ((float4*)tf)[3] = s4[3];
      #pragma unroll
      for (int e = 0; e < 16; ++e) dstA[e] = (bf16)tf[e];
    }
    // B tile: always async (bf16 weights, N multiple of 128 -> no OOR)
    const bf16* sB = p.Bt + (size_t)(n0 + lr) * p.K + k0 + kh;
    bf16* dstB = &Bs[buf][lr * 40 + kh];
    async_ld_b128(dstB, sB);
    async_ld_b128(dstB + 8, sB + 8);
  };

  const int ntiles = p.K / 32;
  stage(0, 0);

  for (int t = 0; t < ntiles; ++t) {
    const int buf = t & 1;
    const bool more = (t + 1) < ntiles;
    if (more) stage(t + 1, buf ^ 1);          // prefetch next tile (other buffer)
    if (more) wait_async<OPS>();              // tile t landed; tile t+1 may fly
    else      wait_async<0>();
    __syncthreads();                           // cross-wave visibility (+DScnt)

    v16bf fa[2];
    #pragma unroll
    for (int i = 0; i < 2; ++i) {
      const bf16* base = &As[buf][(wm + 16*i + lrow) * 40];
      v8bf lo = *(const v8bf*)(base + lhalf * 8);
      v8bf hi = *(const v8bf*)(base + 16 + lhalf * 8);
      fa[i] = __builtin_shufflevector(lo, hi, 0,1,2,3,4,5,6,7,8,9,10,11,12,13,14,15);
    }
    v16bf fb[4];
    #pragma unroll
    for (int j = 0; j < 4; ++j) {
      const bf16* base = &Bs[buf][(wn + 16*j + lrow) * 40];
      v8bf lo = *(const v8bf*)(base + lhalf * 8);
      v8bf hi = *(const v8bf*)(base + 16 + lhalf * 8);
      fb[j] = __builtin_shufflevector(lo, hi, 0,1,2,3,4,5,6,7,8,9,10,11,12,13,14,15);
    }
    #pragma unroll
    for (int i = 0; i < 2; ++i)
      #pragma unroll
      for (int j = 0; j < 4; ++j)
        acc[i][j] = wmma_bf16(fa[i], fb[j], acc[i][j]);
    __syncthreads();                           // buffer free for stage(t+2)
  }

  // ---- epilogue ----
  if (!DOT) {
    #pragma unroll
    for (int i = 0; i < 2; ++i) {
      #pragma unroll
      for (int j = 0; j < 4; ++j) {
        const int col = n0 + wn + 16*j + lrow;
        const float bv = p.bias ? p.bias[col] : 0.f;
        #pragma unroll
        for (int r = 0; r < 8; ++r) {
          const int row = m0 + wm + 16*i + 8*lhalf + r;
          if (row < p.M) {
            float v = acc[i][j][r] + bv;
            if (p.relu) v = fmaxf(v, 0.f);
            if (p.outf) p.outf[(size_t)row * p.N + col] = v;
            if (p.outb) p.outb[(size_t)row * p.N + col] = (bf16)v;
            if (p.outw && col >= p.wlo && col < p.whi)
              p.outw[(size_t)row * p.wld + (col - p.wlo)] = v;
          }
        }
      }
    }
  } else {
    // scores[row] += sum_col (acc+bias) * prevp[cidx[row]][col]  (partial, this N-tile)
    #pragma unroll
    for (int i = 0; i < 2; ++i) {
      #pragma unroll
      for (int r = 0; r < 8; ++r) {
        const int row = m0 + wm + 16*i + 8*lhalf + r;
        const bool valid = row < p.M;
        float s = 0.f;
        if (valid) {
          const int ci = p.cidx[row];
          const float* pp = p.prevp + (size_t)ci * p.N;
          #pragma unroll
          for (int j = 0; j < 4; ++j) {
            const int col = n0 + wn + 16*j + lrow;
            const float v = acc[i][j][r] + p.bias[col];
            s += v * pp[col];
          }
        }
        s += __shfl_xor(s, 1);
        s += __shfl_xor(s, 2);
        s += __shfl_xor(s, 4);
        s += __shfl_xor(s, 8);
        if (valid && lrow == 0) atomicAdd(&p.scores[row], s);
      }
    }
  }
}

// ---------------------------------------------------------------------------
// Support kernels
// ---------------------------------------------------------------------------
__global__ void k_transpose_w(const float* __restrict__ src, bf16* __restrict__ dst,
                              int K, int N) {
  long i = (long)blockIdx.x * blockDim.x + threadIdx.x;
  if (i < (long)K * N) {
    int k = (int)(i / N), n = (int)(i % N);
    dst[(size_t)n * K + k] = (bf16)src[i];
  }
}

__global__ void k_fill_u32(unsigned* p, unsigned v, int n) {
  int i = blockIdx.x * blockDim.x + threadIdx.x;
  if (i < n) p[i] = v;
}

__global__ void k_candidx(const int* __restrict__ edges, int* __restrict__ cidx, int n2, int t) {
  int r = blockIdx.x * blockDim.x + threadIdx.x;
  if (r < n2) cidx[r] = (r < t) ? edges[2*r] : edges[2*(r - t) + 1];
}

__global__ void k_rowdot(const float* __restrict__ A, const float* __restrict__ v,
                         float* __restrict__ out, int M, int N) {
  int row = blockIdx.x * (blockDim.x >> 5) + (threadIdx.x >> 5);
  int lane = threadIdx.x & 31;
  if (row >= M) return;
  const float* a = A + (size_t)row * N;
  float s = 0.f;
  for (int c = lane; c < N; c += 32) s += a[c] * v[c];
  s += __shfl_xor(s, 1);  s += __shfl_xor(s, 2);  s += __shfl_xor(s, 4);
  s += __shfl_xor(s, 8);  s += __shfl_xor(s, 16);
  if (lane == 0) out[row] = s;
}

__global__ void k_segmax(const float* __restrict__ scores, const int* __restrict__ cidx,
                         unsigned* __restrict__ segm, int n) {
  int r = blockIdx.x * blockDim.x + threadIdx.x;
  if (r < n) {
    float x = scores[r];
    unsigned u = __float_as_uint(x);
    unsigned e = (x >= 0.f) ? (u | 0x80000000u) : ~u;   // order-preserving encode
    atomicMax(&segm[cidx[r]], e);
  }
}

__global__ void k_combine(const unsigned* __restrict__ segm, const float* __restrict__ zs,
                          float* __restrict__ mArr, float* __restrict__ denom, int n) {
  int i = blockIdx.x * blockDim.x + threadIdx.x;
  if (i < n) {
    unsigned e = segm[i];
    unsigned u = (e & 0x80000000u) ? (e & 0x7FFFFFFFu) : ~e;
    float m = fmaxf(zs[i], __uint_as_float(u));
    mArr[i] = m;
    denom[i] = __expf(zs[i] - m);       // exp(zero_scores - m) seed
  }
}

__global__ void k_pool(const bf16* __restrict__ newt, const int* __restrict__ cidx,
                       const float* __restrict__ scores, const float* __restrict__ mArr,
                       float* __restrict__ pooled, float* __restrict__ denom,
                       int t, int ldnt, int candoff) {
  int r = blockIdx.x;
  int ci = cidx[r];
  float w = __expf(scores[r] - mArr[ci]);
  const bf16* src = (r < t) ? (newt + (size_t)r * ldnt)
                            : (newt + (size_t)(r - t) * ldnt + candoff);
  for (int c = threadIdx.x; c < H_N; c += blockDim.x)
    atomicAdd(&pooled[(size_t)ci * H_N + c], w * (float)src[c]);
  if (threadIdx.x == 0) atomicAdd(&denom[ci], w);
}

__global__ void k_pool_finalize(float* __restrict__ pooled, const float* __restrict__ denom,
                                size_t n) {
  size_t i = (size_t)blockIdx.x * blockDim.x + threadIdx.x;
  if (i < n) pooled[i] /= denom[i >> 9];   // H_N == 512
}

// ---------------------------------------------------------------------------
// Launch
// ---------------------------------------------------------------------------
static inline int cdiv(long a, long b) { return (int)((a + b - 1) / b); }

template<int AMODE, bool DOT>
static void launch_gemm(const GemmParams& p, hipStream_t st) {
  dim3 grid(p.N / 128, cdiv(p.M, 128));
  wmma_gemm<AMODE, DOT><<<grid, 256, 0, st>>>(p);
}

extern "C" void kernel_launch(void* const* d_in, const int* in_sizes, int n_in,
                              void* d_out, int out_size, void* d_ws, size_t ws_size,
                              hipStream_t stream) {
  (void)in_sizes; (void)n_in; (void)out_size; (void)ws_size;
  const float* obj_vecs  = (const float*)d_in[0];
  const float* pred_vecs = (const float*)d_in[1];
  const int*   edges     = (const int*)  d_in[2];
  const float* n1_w1 = (const float*)d_in[3];
  const float* n1_b1 = (const float*)d_in[4];
  const float* n1_w2 = (const float*)d_in[5];
  const float* n1_b2 = (const float*)d_in[6];
  const float* n2_w1 = (const float*)d_in[7];
  const float* n2_b1 = (const float*)d_in[8];
  const float* n2_w2 = (const float*)d_in[9];
  const float* n2_b2 = (const float*)d_in[10];
  const float* proj_w = (const float*)d_in[11];
  const float* proj_b = (const float*)d_in[12];
  const float* sim_w  = (const float*)d_in[13];
  const float* sim_b  = (const float*)d_in[14];
  float* out = (float*)d_out;

  char* ws = (char*)d_ws;
  size_t off = 0;
  auto alloc = [&](size_t bytes) -> char* {
    char* p = ws + off;
    off = (off + bytes + 255) & ~(size_t)255;
    return p;
  };

  bf16* w1t    = (bf16*)alloc((size_t)H_N  * K1_N * 2);
  bf16* w2t    = (bf16*)alloc((size_t)N2_N * H_N  * 2);
  bf16* n2w1t  = (bf16*)alloc((size_t)H_N  * H_N  * 2);
  bf16* n2w2t  = (bf16*)alloc((size_t)DOUT_N * H_N * 2);
  bf16* projwt = (bf16*)alloc((size_t)H_N  * DIN_N * 2);
  bf16* simwt  = (bf16*)alloc((size_t)H_N  * H_N  * 2);
  int*      cidx   = (int*)alloc((size_t)2 * T_N * 4);
  float*    scores = (float*)alloc((size_t)2 * T_N * 4);
  float*    zs     = (float*)alloc((size_t)O_N * 4);
  unsigned* segm   = (unsigned*)alloc((size_t)O_N * 4);
  float*    mArr   = (float*)alloc((size_t)O_N * 4);
  float*    denom  = (float*)alloc((size_t)O_N * 4);
  bf16*     newt   = (bf16*)alloc((size_t)T_N * N2_N * 2);           // 230 MB
  float*    pooled = (float*)alloc((size_t)O_N * H_N * 4);           // 102 MB
  char*     big1   = alloc((size_t)O_N * H_N * 4);                   // 102 MB shared
  bf16*     hbuf   = (bf16*)big1;        // h (T,512) bf16 — dead after GEMM2
  float*    prevp  = (float*)big1;       // prev_proj (O,512) f32 — written after
  char*     big2   = alloc((size_t)O_N * H_N * 2);                   // 51 MB shared
  bf16*     prevb  = (bf16*)big2;        // prev (O,512) bf16 — dead after GEMM4
  bf16*     h2b    = (bf16*)big2;        // net2 hidden — written after

  const int thr = 256;

  // weights -> bf16 transposed (N,K)
  k_transpose_w<<<cdiv((long)K1_N*H_N, thr), thr, 0, stream>>>(n1_w1, w1t, K1_N, H_N);
  k_transpose_w<<<cdiv((long)H_N*N2_N, thr), thr, 0, stream>>>(n1_w2, w2t, H_N, N2_N);
  k_transpose_w<<<cdiv((long)H_N*H_N,  thr), thr, 0, stream>>>(n2_w1, n2w1t, H_N, H_N);
  k_transpose_w<<<cdiv((long)H_N*DOUT_N, thr), thr, 0, stream>>>(n2_w2, n2w2t, H_N, DOUT_N);
  k_transpose_w<<<cdiv((long)DIN_N*H_N, thr), thr, 0, stream>>>(proj_w, projwt, DIN_N, H_N);
  k_transpose_w<<<cdiv((long)H_N*H_N,  thr), thr, 0, stream>>>(sim_w, simwt, H_N, H_N);
  k_candidx<<<cdiv(2*T_N, thr), thr, 0, stream>>>(edges, cidx, 2*T_N, T_N);

  // GEMM1: h = relu(t_in @ n1_w1 + b)   (gather-concat fused, AMODE 1)
  {
    GemmParams p{};
    p.obj = obj_vecs; p.pred = pred_vecs; p.edges = edges;
    p.Bt = w1t; p.bias = n1_b1; p.relu = 1;
    p.M = T_N; p.N = H_N; p.K = K1_N; p.lda = 0;
    p.outb = hbuf;
    launch_gemm<1, false>(p, stream);
  }
  // GEMM2: new_t = relu(h @ n1_w2 + b); new_p f32 window -> d_out[O*128 ...]
  {
    GemmParams p{};
    p.A = hbuf; p.lda = H_N;
    p.Bt = w2t; p.bias = n1_b2; p.relu = 1;
    p.M = T_N; p.N = N2_N; p.K = H_N;
    p.outb = newt;
    p.outw = out + (size_t)O_N * DOUT_N; p.wlo = H_N; p.whi = H_N + DOUT_N; p.wld = DOUT_N;
    launch_gemm<0, false>(p, stream);
  }
  // GEMM3: prev = obj_vecs @ proj_w + b   (AMODE 3, f32 input)
  {
    GemmParams p{};
    p.Af = obj_vecs; p.lda = DIN_N;
    p.Bt = projwt; p.bias = proj_b; p.relu = 0;
    p.M = O_N; p.N = H_N; p.K = DIN_N;
    p.outb = prevb;
    launch_gemm<3, false>(p, stream);
  }
  // GEMM4: prev_proj = prev @ sim_w + b (f32 out, overwrites hbuf region)
  {
    GemmParams p{};
    p.A = prevb; p.lda = H_N;
    p.Bt = simwt; p.bias = sim_b; p.relu = 0;
    p.M = O_N; p.N = H_N; p.K = H_N;
    p.outf = prevp;
    launch_gemm<0, false>(p, stream);
  }
  // zero_scores = prev_proj @ sim_b
  k_rowdot<<<cdiv(O_N, 8), 256, 0, stream>>>(prevp, sim_b, zs, O_N, H_N);

  hipMemsetAsync(scores, 0, (size_t)2 * T_N * 4, stream);
  k_fill_u32<<<cdiv(O_N, thr), thr, 0, stream>>>(segm, 0x007FFFFFu, O_N);  // enc(-inf)

  // GEMM5 (DOT): scores[r] = (cand @ sim_w + b) . prev_proj[cidx[r]]   (AMODE 2)
  {
    GemmParams p{};
    p.A = newt; p.lda = N2_N; p.tsplit = T_N; p.candoff = H_N + DOUT_N;
    p.Bt = simwt; p.bias = sim_b; p.relu = 0;
    p.M = 2 * T_N; p.N = H_N; p.K = H_N;
    p.prevp = prevp; p.cidx = cidx; p.scores = scores;
    launch_gemm<2, true>(p, stream);
  }

  k_segmax<<<cdiv(2*T_N, thr), thr, 0, stream>>>(scores, cidx, segm, 2*T_N);
  k_combine<<<cdiv(O_N, thr), thr, 0, stream>>>(segm, zs, mArr, denom, O_N);
  hipMemsetAsync(pooled, 0, (size_t)O_N * H_N * 4, stream);
  k_pool<<<2*T_N, 256, 0, stream>>>(newt, cidx, scores, mArr, pooled, denom,
                                    T_N, N2_N, H_N + DOUT_N);
  k_pool_finalize<<<cdiv((long)O_N*H_N, thr), thr, 0, stream>>>(pooled, denom, (size_t)O_N*H_N);

  // GEMM6: h2 = relu(pooled @ n2_w1 + b)   (AMODE 3, f32 pooled)
  {
    GemmParams p{};
    p.Af = pooled; p.lda = H_N;
    p.Bt = n2w1t; p.bias = n2_b1; p.relu = 1;
    p.M = O_N; p.N = H_N; p.K = H_N;
    p.outb = h2b;
    launch_gemm<3, false>(p, stream);
  }
  // GEMM7: new_obj = relu(h2 @ n2_w2 + b) -> d_out[0 : O*128] (f32)
  {
    GemmParams p{};
    p.A = h2b; p.lda = H_N;
    p.Bt = n2w2t; p.bias = n2_b2; p.relu = 1;
    p.M = O_N; p.N = DOUT_N; p.K = H_N;
    p.outf = out;
    launch_gemm<0, false>(p, stream);
  }
}
